// MultiHeadAttention_17867063951739
// MI455X (gfx1250) — matrix operand order repaired and checked by reading, not compiled, not run on hardware
//
#include <hip/hip_runtime.h>
#include <hip/hip_bf16.h>

#define SEQ 4096
#define DM  1024
#define NH  16
#define DK  64

typedef __attribute__((ext_vector_type(16))) __bf16 v16bf;
typedef __attribute__((ext_vector_type(8)))  __bf16 v8bf;
typedef __attribute__((ext_vector_type(8)))  float  v8f;

__device__ __forceinline__ __bf16 f2bf(float x) {
  unsigned u = __builtin_bit_cast(unsigned, x);
  unsigned r = u + 0x7FFFu + ((u >> 16) & 1u);   // round-to-nearest-even
  unsigned short h = (unsigned short)(r >> 16);
  return __builtin_bit_cast(__bf16, h);
}
__device__ __forceinline__ float bf2f(__bf16 x) {
  unsigned short h = __builtin_bit_cast(unsigned short, x);
  unsigned u = ((unsigned)h) << 16;
  return __builtin_bit_cast(float, u);
}

// A-fragment (16x32, 16-bit): lane L holds row M=L&15, K in {kh..kh+7} u {16+kh..16+kh+7}, kh=8*(L>>4)
__device__ __forceinline__ v16bf load_a_frag(const __bf16* __restrict__ A, int lda,
                                             int row0, int k0, int lane) {
  const int m  = row0 + (lane & 15);
  const int kh = (lane >> 4) << 3;
  const __bf16* p = A + (size_t)m * lda + k0 + kh;
  v8bf lo = *(const v8bf*)p;
  v8bf hi = *(const v8bf*)(p + 16);
  return __builtin_shufflevector(lo, hi, 0,1,2,3,4,5,6,7,8,9,10,11,12,13,14,15);
}

// B-fragment (32x16, 16-bit), B[k][n] = W[n][k] (row-major W): lane L holds col N=L&15,
// K in {kb..kb+7} u {16+kb..16+kb+7}, kb = 8*(L>>4)  ->  two 16B loads from W's row.
__device__ __forceinline__ v16bf load_b_frag(const __bf16* __restrict__ W, int ldw,
                                             int col0, int k0, int lane) {
  const int n  = col0 + (lane & 15);
  const int kb = k0 + ((lane >> 4) << 3);
  const __bf16* p = W + (size_t)n * ldw + kb;
  v8bf lo = *(const v8bf*)p;
  v8bf hi = *(const v8bf*)(p + 16);
  return __builtin_shufflevector(lo, hi, 0,1,2,3,4,5,6,7,8,9,10,11,12,13,14,15);
}

__global__ void k_f32_to_bf16(const float* __restrict__ src, __bf16* __restrict__ dst, int n) {
  int i = blockIdx.x * blockDim.x + threadIdx.x;
  if (i < n) dst[i] = f2bf(src[i]);
}

// C[m,n] = sum_k A[m,k]*W[n,k] + bias[n].  One wave -> 32x64 tile of C (8 WMMA / k-step).
__global__ void k_gemm_bf16(const __bf16* __restrict__ A, const __bf16* __restrict__ W,
                            const float* __restrict__ bias,
                            __bf16* __restrict__ Cb, float* __restrict__ Cf,
                            int M, int N, int Kd) {
  const int lane = threadIdx.x & 31;
  const int wid  = blockIdx.x * (blockDim.x >> 5) + (threadIdx.x >> 5);
  const int ntiles = N >> 6;
  const int total  = (M >> 5) * ntiles;
  if (wid >= total) return;
  const int row0 = (wid / ntiles) << 5;
  const int col0 = (wid % ntiles) << 6;

  v8f acc[2][4] = {};
  for (int k0 = 0; k0 < Kd; k0 += 32) {
    v16bf a0 = load_a_frag(A, Kd, row0,      k0, lane);
    v16bf a1 = load_a_frag(A, Kd, row0 + 16, k0, lane);
#pragma unroll
    for (int t = 0; t < 4; ++t) {
      v16bf b = load_b_frag(W, Kd, col0 + (t << 4), k0, lane);
      acc[0][t] = __builtin_amdgcn_wmma_f32_16x16x32_bf16(false, a0, false, b,
                                                          (short)0, acc[0][t], false, false);
      acc[1][t] = __builtin_amdgcn_wmma_f32_16x16x32_bf16(false, a1, false, b,
                                                          (short)0, acc[1][t], false, false);
    }
  }
#pragma unroll
  for (int rt = 0; rt < 2; ++rt) {
    const int rbase = row0 + (rt << 4) + ((lane >> 4) << 3);
#pragma unroll
    for (int t = 0; t < 4; ++t) {
      const int col = col0 + (t << 4) + (lane & 15);
      const float bv = bias ? bias[col] : 0.0f;
#pragma unroll
      for (int r = 0; r < 8; ++r) {
        float v = acc[rt][t][r] + bv;
        size_t idx = (size_t)(rbase + r) * N + col;
        if (Cf) Cf[idx] = v;
        if (Cb) Cb[idx] = f2bf(v);
      }
    }
  }
}

// Pass 1: per-column (key m) max/sum-exp of S[n,m]=scale*Qh[n,:]·Kh[m,:], online over n.
// One wave -> 32 key columns (Q fragments feed 2 WMMAs each).
__global__ void k_attn_stats(const __bf16* __restrict__ Q, const __bf16* __restrict__ K,
                             float* __restrict__ colmax, float* __restrict__ colsum) {
  const int lane = threadIdx.x & 31;
  const int task = blockIdx.x * (blockDim.x >> 5) + (threadIdx.x >> 5);
  const int h  = task >> 7;          // 128 m-tiles (32 cols) per head
  const int m0 = (task & 127) << 5;
  const float scale = 0.125f;        // 1/sqrt(64)

  const __bf16* Kh = K + h * DK;
  const __bf16* Qh = Q + h * DK;
  v16bf bk[2][2];
#pragma unroll
  for (int mt = 0; mt < 2; ++mt) {
    bk[mt][0] = load_b_frag(Kh, DM, m0 + (mt << 4), 0,  lane);
    bk[mt][1] = load_b_frag(Kh, DM, m0 + (mt << 4), 32, lane);
  }

  float mx[2] = {-3.0e38f, -3.0e38f};
  float sm[2] = {0.0f, 0.0f};
  for (int nt = 0; nt < SEQ; nt += 16) {
    v16bf a0 = load_a_frag(Qh, DM, nt, 0,  lane);
    v16bf a1 = load_a_frag(Qh, DM, nt, 32, lane);
#pragma unroll
    for (int mt = 0; mt < 2; ++mt) {
      v8f s = {};
      s = __builtin_amdgcn_wmma_f32_16x16x32_bf16(false, a0, false, bk[mt][0], (short)0, s, false, false);
      s = __builtin_amdgcn_wmma_f32_16x16x32_bf16(false, a1, false, bk[mt][1], (short)0, s, false, false);
      float tm = -3.0e38f;
#pragma unroll
      for (int r = 0; r < 8; ++r) tm = fmaxf(tm, s[r] * scale);
      float mn  = fmaxf(mx[mt], tm);
      float acc = sm[mt] * __expf(mx[mt] - mn);
#pragma unroll
      for (int r = 0; r < 8; ++r) acc += __expf(s[r] * scale - mn);
      sm[mt] = acc; mx[mt] = mn;
    }
  }
  // merge the two half-wave row-subsets of the same column
#pragma unroll
  for (int mt = 0; mt < 2; ++mt) {
    float mo = __shfl_xor(mx[mt], 16, 32);
    float so = __shfl_xor(sm[mt], 16, 32);
    float mn = fmaxf(mx[mt], mo);
    float sv = sm[mt] * __expf(mx[mt] - mn) + so * __expf(mo - mn);
    if (lane < 16) {
      colmax[h * SEQ + m0 + (mt << 4) + lane] = mn;
      colsum[h * SEQ + m0 + (mt << 4) + lane] = sv;
    }
  }
}

// Vt[h][d][m] = V[m, h*64+d] / colsum[h][m]  (bf16, transposed for contiguous B-frag loads)
__global__ void k_build_vt(const __bf16* __restrict__ V, const float* __restrict__ colsum,
                           __bf16* __restrict__ Vt) {
  int i = blockIdx.x * blockDim.x + threadIdx.x;   // NH*DK*SEQ = 2^22
  if (i >= NH * DK * SEQ) return;
  int m = i & (SEQ - 1);
  int d = (i >> 12) & (DK - 1);
  int h = i >> 18;
  float inv = 1.0f / colsum[h * SEQ + m];
  Vt[i] = f2bf(bf2f(V[(size_t)m * DM + h * DK + d]) * inv);
}

// Pass 2: O[n, h*64+d] = sum_m exp(S[n,m]-colmax[m]) * Vt[h][d][m].
// One wave -> 32n x 64d (K and Vt B-fragments each feed 2 WMMAs).
__global__ void k_attn_out(const __bf16* __restrict__ Q, const __bf16* __restrict__ K,
                           const __bf16* __restrict__ Vt, const float* __restrict__ colmax,
                           __bf16* __restrict__ H) {
  __shared__ float sS[4][32][33];
  const int lane = threadIdx.x & 31;
  const int w    = threadIdx.x >> 5;
  const int task = blockIdx.x * (blockDim.x >> 5) + w;
  const int h  = task >> 7;          // 128 n-tiles (32 rows) per head
  const int n0 = (task & 127) << 5;
  const float scale = 0.125f;

  const __bf16* Qh = Q + h * DK;
  const __bf16* Kh = K + h * DK;
  const __bf16* Vh = Vt + (size_t)h * DK * SEQ;
  const float*  cmax = colmax + h * SEQ;

  v16bf a0[2], a1[2];
#pragma unroll
  for (int nt = 0; nt < 2; ++nt) {
    a0[nt] = load_a_frag(Qh, DM, n0 + (nt << 4), 0,  lane);
    a1[nt] = load_a_frag(Qh, DM, n0 + (nt << 4), 32, lane);
  }

  v8f acc[2][4] = {};
  const int nr   = lane & 15;
  const int kh   = (lane >> 4) << 3;
  const int srow = (lane >> 4) << 3;

  for (int m0 = 0; m0 < SEQ; m0 += 32) {
    // S tile (32n x 32m) via WMMA, parked in LDS in [n][m] order
#pragma unroll
    for (int mt = 0; mt < 2; ++mt) {
      v16bf bk0 = load_b_frag(Kh, DM, m0 + (mt << 4), 0,  lane);
      v16bf bk1 = load_b_frag(Kh, DM, m0 + (mt << 4), 32, lane);
      const int c = (mt << 4) + (lane & 15);
#pragma unroll
      for (int nt = 0; nt < 2; ++nt) {
        v8f s = {};
        s = __builtin_amdgcn_wmma_f32_16x16x32_bf16(false, a0[nt], false, bk0, (short)0, s, false, false);
        s = __builtin_amdgcn_wmma_f32_16x16x32_bf16(false, a1[nt], false, bk1, (short)0, s, false, false);
#pragma unroll
        for (int r = 0; r < 8; ++r) sS[w][(nt << 4) + srow + r][c] = s[r] * scale;
      }
    }
    // column maxima for this 32-key chunk, hoisted once
    float cm0[8], cm1[8];
#pragma unroll
    for (int j = 0; j < 8; ++j) {
      cm0[j] = cmax[m0 + kh + j];
      cm1[j] = cmax[m0 + 16 + kh + j];
    }
    // cross-lane LDS dependency: make writes visible before re-reading in A-frag order
    asm volatile("s_wait_dscnt 0" ::: "memory");
    v16bf af[2];
#pragma unroll
    for (int nt = 0; nt < 2; ++nt) {
#pragma unroll
      for (int j = 0; j < 8; ++j) {
        float p0 = __expf(sS[w][(nt << 4) + nr][kh + j]      - cm0[j]);
        float p1 = __expf(sS[w][(nt << 4) + nr][16 + kh + j] - cm1[j]);
        af[nt][j]     = f2bf(p0);
        af[nt][j + 8] = f2bf(p1);
      }
    }
#pragma unroll
    for (int t = 0; t < 4; ++t) {
      v16bf bv = load_b_frag(Vh, SEQ, t << 4, m0, lane);
      acc[0][t] = __builtin_amdgcn_wmma_f32_16x16x32_bf16(false, af[0], false, bv,
                                                          (short)0, acc[0][t], false, false);
      acc[1][t] = __builtin_amdgcn_wmma_f32_16x16x32_bf16(false, af[1], false, bv,
                                                          (short)0, acc[1][t], false, false);
    }
  }
#pragma unroll
  for (int nt = 0; nt < 2; ++nt) {
    const int rb = n0 + (nt << 4) + ((lane >> 4) << 3);
#pragma unroll
    for (int t = 0; t < 4; ++t) {
      const int col = h * DK + (t << 4) + (lane & 15);
#pragma unroll
      for (int r = 0; r < 8; ++r)
        H[(size_t)(rb + r) * DM + col] = f2bf(acc[nt][t][r]);
    }
  }
}

extern "C" void kernel_launch(void* const* d_in, const int* in_sizes, int n_in,
                              void* d_out, int out_size, void* d_ws, size_t ws_size,
                              hipStream_t stream) {
  const float* E   = (const float*)d_in[0];
  const float* W_Q = (const float*)d_in[1];
  const float* b_Q = (const float*)d_in[2];
  const float* W_K = (const float*)d_in[3];
  const float* b_K = (const float*)d_in[4];
  const float* W_V = (const float*)d_in[5];
  const float* b_V = (const float*)d_in[6];
  const float* W_O = (const float*)d_in[7];
  const float* b_O = (const float*)d_in[8];
  float* out = (float*)d_out;
  (void)in_sizes; (void)n_in; (void)out_size; (void)ws_size;

  char* ws = (char*)d_ws;
  size_t off = 0;
  auto take = [&](size_t bytes) -> void* {
    void* p = ws + off;
    off += (bytes + 255) & ~(size_t)255;
    return p;
  };
  __bf16* Ebf = (__bf16*)take((size_t)SEQ * DM * 2);
  __bf16* Wq  = (__bf16*)take((size_t)DM * DM * 2);
  __bf16* Wk  = (__bf16*)take((size_t)DM * DM * 2);
  __bf16* Wv  = (__bf16*)take((size_t)DM * DM * 2);
  __bf16* Wo  = (__bf16*)take((size_t)DM * DM * 2);
  __bf16* Qb  = (__bf16*)take((size_t)SEQ * DM * 2);
  __bf16* Kb  = (__bf16*)take((size_t)SEQ * DM * 2);
  __bf16* Vb  = (__bf16*)take((size_t)SEQ * DM * 2);
  __bf16* Vt  = (__bf16*)take((size_t)NH * DK * SEQ * 2);
  __bf16* Hb  = (__bf16*)take((size_t)SEQ * DM * 2);
  float* colmax = (float*)take((size_t)NH * SEQ * 4);
  float* colsum = (float*)take((size_t)NH * SEQ * 4);

  // fp32 -> bf16 staging
  k_f32_to_bf16<<<(SEQ * DM) / 256, 256, 0, stream>>>(E,   Ebf, SEQ * DM);
  k_f32_to_bf16<<<(DM * DM) / 256, 256, 0, stream>>>(W_Q, Wq, DM * DM);
  k_f32_to_bf16<<<(DM * DM) / 256, 256, 0, stream>>>(W_K, Wk, DM * DM);
  k_f32_to_bf16<<<(DM * DM) / 256, 256, 0, stream>>>(W_V, Wv, DM * DM);
  k_f32_to_bf16<<<(DM * DM) / 256, 256, 0, stream>>>(W_O, Wo, DM * DM);

  const int gemm_blocks = ((SEQ / 32) * (DM / 64)) / 8;   // 256 blocks x 8 waves
  k_gemm_bf16<<<gemm_blocks, 256, 0, stream>>>(Ebf, Wq, b_Q, Qb, nullptr, SEQ, DM, DM);
  k_gemm_bf16<<<gemm_blocks, 256, 0, stream>>>(Ebf, Wk, b_K, Kb, nullptr, SEQ, DM, DM);
  k_gemm_bf16<<<gemm_blocks, 256, 0, stream>>>(Ebf, Wv, b_V, Vb, nullptr, SEQ, DM, DM);

  const int attn_blocks = (NH * (SEQ / 32)) / 4;          // 512 blocks x 4 waves
  k_attn_stats<<<attn_blocks, 128, 0, stream>>>(Qb, Kb, colmax, colsum);
  k_build_vt<<<(NH * DK * SEQ) / 256, 256, 0, stream>>>(Vb, colsum, Vt);
  k_attn_out<<<attn_blocks, 128, 0, stream>>>(Qb, Kb, Vt, colmax, Hb);

  k_gemm_bf16<<<gemm_blocks, 256, 0, stream>>>(Hb, Wo, b_O, nullptr, out, SEQ, DM, DM);
}
